// TemporalInteractionNet_30666066493880
// MI455X (gfx1250) — matrix operand order repaired and checked by reading, not compiled, not run on hardware
//
#include <hip/hip_runtime.h>
#include <hip/hip_bf16.h>
#include <math.h>
#include <stdint.h>

#define SEQ  16
#define NND  1024
#define FIN  64
#define HH   128
#define OUTC 32
#define EE   16384
#define CDIV(a,b) (((a)+(b)-1)/(b))

typedef __attribute__((ext_vector_type(16))) _Float16 v16h;
typedef __attribute__((ext_vector_type(8)))  float    v8f;

__device__ __forceinline__ float sigmoidf_(float x) { return 1.f / (1.f + __expf(-x)); }

// Async global->LDS DMA (CDNA5): per-lane 16B, tracked by ASYNCcnt.
// lds_off: 32-bit LDS byte address (VDST VGPR); gaddr: 64-bit global address (GV mode).
__device__ __forceinline__ void async_b128(unsigned lds_off, const float* gaddr) {
  asm volatile("global_load_async_to_lds_b128 %0, %1, off"
               :: "v"(lds_off), "v"(gaddr) : "memory");
}
__device__ __forceinline__ void wait_async0() {
  asm volatile("s_wait_asynccnt 0x0" ::: "memory");
}
__device__ __forceinline__ unsigned lds_off_of(const void* p) {
  return (unsigned)(uintptr_t)p;   // low 32 bits of flat addr = LDS byte address
}

// ---------------------------------------------------------------------------
// Generic WMMA GEMM: D[M,N] = act(alpha * A[M,K] @ W[N,K]^T + bias + addsrc)
// One wave (32 threads) per 16x16 output tile. All dims multiples of 16 (K of 32).
// Tiles staged raw-f32 via GLOBAL_LOAD_ASYNC_TO_LDS_B128; f32->f16 at fragment build.
// act: 0=none 1=sigmoid 2=tanh 3=relu
// ---------------------------------------------------------------------------
__global__ void gemm_wmma_kernel(const float* __restrict__ A, int lda,
                                 const float* __restrict__ W, int ldw,
                                 const float* __restrict__ bias,
                                 const float* __restrict__ addsrc, int ldadd,
                                 float* __restrict__ D, int ldd,
                                 int K, float alpha, int act) {
  __shared__ float As32[16][32];
  __shared__ float Bs32[16][32];
  const int lane = threadIdx.x;
  const int m0 = blockIdx.y * 16;
  const int n0 = blockIdx.x * 16;
  const int half = lane >> 4;
  const int r = lane & 15;
  const int ofs = half * 8;
  const int cb = half * 16;                       // this lane's 16-float row-half
  const unsigned as_b = lds_off_of(&As32[r][cb]);
  const unsigned bs_b = lds_off_of(&Bs32[r][cb]);
  v8f acc = {};
  for (int k0 = 0; k0 < K; k0 += 32) {
    const float* arow = A + (long)(m0 + r) * lda + k0 + cb;
    const float* wrow = W + (long)(n0 + r) * ldw + k0 + cb;
    #pragma unroll
    for (int j = 0; j < 4; ++j) {                 // 4 x 16B per tile per lane
      async_b128(as_b + j * 16, arow + j * 4);
      async_b128(bs_b + j * 16, wrow + j * 4);
    }
    wait_async0();                                // single-wave block: counter wait suffices
    __syncthreads();
    v16h af, bf;
    #pragma unroll
    for (int e = 0; e < 16; ++e)                  // A: contiguous 16 K per half-wave
      af[e] = (_Float16)As32[r][cb + e];
    #pragma unroll
    for (int e = 0; e < 16; ++e)                  // B: contiguous 16 K per half-wave
      bf[e] = (_Float16)Bs32[r][cb + e];
    acc = __builtin_amdgcn_wmma_f32_16x16x32_f16(false, af, false, bf,
                                                 (short)0, acc, false, false);
    __syncthreads();
  }
  const int col = n0 + r;
  #pragma unroll
  for (int rr = 0; rr < 8; ++rr) {
    const int row = m0 + half * 8 + rr;
    float v = acc[rr] * alpha;
    if (bias)   v += bias[col];
    if (addsrc) v += addsrc[(long)row * ldadd + col];
    if (act == 1)      v = sigmoidf_(v);
    else if (act == 2) v = tanhf(v);
    else if (act == 3) v = fmaxf(v, 0.f);
    D[(long)row * ldd + col] = v;
  }
}

static inline void gemm(hipStream_t s, const float* A, int lda, const float* W, int ldw,
                        const float* bias, const float* add, int ldadd,
                        float* D, int ldd, int M, int N, int K,
                        float alpha = 1.f, int act = 0) {
  dim3 g(N / 16, M / 16);
  gemm_wmma_kernel<<<g, 32, 0, s>>>(A, lda, W, ldw, bias, add, ldadd, D, ldd, K, alpha, act);
}

// ---------------------------------------------------------------------------
// Pairwise interaction: inter[i,j] = sigmoid(relu(relu(za_i+zb_j+b1)@W2^T+b2)@w3+b3)
// grid = (N/16 j-tiles, N i). One wave computes a 16(j) x 64 h2 tile via 16 WMMAs,
// then fuses the w3 dot + sigmoid with half-wave shuffles. zb tile staged async.
// ---------------------------------------------------------------------------
__global__ void k_pairwise(const float* __restrict__ za, const float* __restrict__ zb,
                           const float* __restrict__ b1, const _Float16* __restrict__ W2,
                           const float* __restrict__ b2, const float* __restrict__ w3,
                           const float* __restrict__ b3, float* __restrict__ inter) {
  __shared__ float combo[HH];
  __shared__ float zbs[16][HH];
  const int lane = threadIdx.x;
  const int i = blockIdx.y;
  const int j0 = blockIdx.x * 16;
  // async-DMA the 16x128 f32 zb tile: each lane owns 64 consecutive floats (16 chunks x 4 floats)
  {
    const int m = lane >> 1;                       // rows 0..15, two lanes per row
    const int c = (lane & 1) * 64;                 // half-row = 64 floats
    unsigned ldsb = lds_off_of(&zbs[m][c]);
    const float* g = zb + (long)(j0 + m) * HH + c;
    #pragma unroll
    for (int j = 0; j < 16; ++j)
      async_b128(ldsb + j * 16, g + j * 4);
  }
  for (int t = lane; t < HH; t += 32) combo[t] = za[(long)i * HH + t] + b1[t];
  wait_async0();
  __syncthreads();
  const int half = lane >> 4;
  const int r = lane & 15;
  const int ofs = half * 8;
  v8f acc[4] = {v8f{}, v8f{}, v8f{}, v8f{}};
  for (int k0 = 0; k0 < HH; k0 += 32) {
    v16h af;
    #pragma unroll
    for (int t = 0; t < 16; ++t)
      af[t] = (_Float16)fmaxf(combo[k0 + half * 16 + t] + zbs[r][k0 + half * 16 + t], 0.f);
    #pragma unroll
    for (int nt = 0; nt < 4; ++nt) {
      const _Float16* w = W2 + (long)(nt * 16 + r) * HH + k0 + half * 16;
      v16h bf;
      #pragma unroll
      for (int e = 0; e < 16; ++e) bf[e] = w[e];
      acc[nt] = __builtin_amdgcn_wmma_f32_16x16x32_f16(false, af, false, bf,
                                                       (short)0, acc[nt], false, false);
    }
  }
  const float bias3 = b3[0];
  #pragma unroll
  for (int rr = 0; rr < 8; ++rr) {
    float p = 0.f;
    #pragma unroll
    for (int nt = 0; nt < 4; ++nt) {
      int c = nt * 16 + r;
      p += fmaxf(acc[nt][rr] + b2[c], 0.f) * w3[c];
    }
    for (int off = 8; off >= 1; off >>= 1) p += __shfl_xor(p, off, 16);
    if (r == 0) {
      int m = half * 8 + rr;
      inter[(long)i * NND + j0 + m] = sigmoidf_(p + bias3);
    }
  }
}

// ------------------------- small helper kernels ----------------------------
__global__ void k_fill(float* p, float v, int n) {
  int i = blockIdx.x * blockDim.x + threadIdx.x;
  if (i < n) p[i] = v;
}
__global__ void k_deg(const int* __restrict__ dst, const float* __restrict__ w,
                      float* deg, int e_cnt) {
  int e = blockIdx.x * blockDim.x + threadIdx.x;
  if (e < e_cnt) atomicAdd(&deg[dst[e]], w[e]);
}
__global__ void k_dis(const float* deg, float* dis, int n) {
  int i = blockIdx.x * blockDim.x + threadIdx.x;
  if (i < n) { float d = deg[i]; dis[i] = d > 0.f ? rsqrtf(fmaxf(d, 1e-12f)) : 0.f; }
}
__global__ void k_norm(const int* src, const int* dst, const float* w,
                       const float* dis, float* nrm, int e_cnt) {
  int e = blockIdx.x * blockDim.x + threadIdx.x;
  if (e < e_cnt) nrm[e] = dis[src[e]] * w[e] * dis[dst[e]];
}
// c[t,n,f] = xw[t,n,f]*dis[n]^2 + bias[f]   (self-loop term + GCN bias)
__global__ void k_gcn_init(const float* __restrict__ xw, const float* __restrict__ dis,
                           const float* __restrict__ bias, float* __restrict__ c, int total) {
  int i = blockIdx.x * blockDim.x + threadIdx.x;
  if (i >= total) return;
  int f = i & (HH - 1);
  int n = (i >> 7) & (NND - 1);
  float d = dis[n];
  c[i] = xw[i] * d * d + bias[f];
}
__global__ void k_gcn_scatter(const float* __restrict__ xw, const int* __restrict__ src,
                              const int* __restrict__ dst, const float* __restrict__ nrm,
                              float* __restrict__ c, int total) {
  int idx = blockIdx.x * blockDim.x + threadIdx.x;
  if (idx >= total) return;                 // total = SEQ*EE*16 (chunks of 8 floats)
  int chunk = idx & 15;
  int e = (idx >> 4) & (EE - 1);
  int t = idx >> 18;                        // 4 + 14
  int s = src[e], d = dst[e];
  float nv = nrm[e];
  const float* xs = xw + ((long)t * NND + s) * HH + chunk * 8;
  float* cd = c + ((long)t * NND + d) * HH + chunk * 8;
  #pragma unroll
  for (int j = 0; j < 8; ++j) atomicAdd(&cd[j], xs[j] * nv);
}
__global__ void k_gates(const float* Z, const float* Hh, float* h, int total) {
  int i = blockIdx.x * blockDim.x + threadIdx.x;
  if (i < total) h[i] = (1.f - Z[i]) * Hh[i];
}
__global__ void k_meantime(const float* __restrict__ hseq, float* __restrict__ hmean, int total) {
  int i = blockIdx.x * blockDim.x + threadIdx.x;
  if (i >= total) return;                   // total = N*H
  int n = i >> 7, f = i & (HH - 1);
  float s = 0.f;
  for (int t = 0; t < SEQ; ++t) s += hseq[((long)t * NND + n) * HH + f];
  hmean[i] = s * (1.f / SEQ);
}
__global__ void k_unpack_conv(const float* __restrict__ Wc, float* Wsum, float* W0b,
                              float* W2b, int total) {
  int i = blockIdx.x * blockDim.x + threadIdx.x;
  if (i >= total) return;                   // total = H*H ; Wc is [H][H][3]
  float w0 = Wc[3 * i], w1 = Wc[3 * i + 1], w2 = Wc[3 * i + 2];
  Wsum[i] = w0 + w1 + w2; W0b[i] = w0; W2b[i] = w2;
}
__global__ void k_scores(const float* __restrict__ q, const float* __restrict__ kk,
                         const int* src, const int* dst, float* sc, int e_cnt) {
  int e = blockIdx.x * blockDim.x + threadIdx.x;
  if (e >= e_cnt) return;
  const float* qr = q + (long)dst[e] * HH;
  const float* kr = kk + (long)src[e] * HH;
  float s = 0.f;
  for (int f = 0; f < HH; ++f) s += qr[f] * kr[f];
  sc[e] = s * 0.08838834764831845f;         // 1/sqrt(128)
}
__device__ void atomicMaxF(float* addr, float val) {
  int* ia = (int*)addr;
  int old = __float_as_int(*addr);
  while (__int_as_float(old) < val) {
    int assumed = old;
    old = atomicCAS(ia, assumed, __float_as_int(val));
    if (old == assumed) break;
  }
}
__global__ void k_segmax(const float* sc, const int* dst, float* mrow, int e_cnt) {
  int e = blockIdx.x * blockDim.x + threadIdx.x;
  if (e < e_cnt) atomicMaxF(&mrow[dst[e]], sc[e]);
}
__global__ void k_expden(const float* sc, const int* dst, const float* mrow,
                         float* ex, float* den, int e_cnt) {
  int e = blockIdx.x * blockDim.x + threadIdx.x;
  if (e >= e_cnt) return;
  float v = __expf(sc[e] - mrow[dst[e]]);
  ex[e] = v;
  atomicAdd(&den[dst[e]], v);
}
__global__ void k_attsc(const float* __restrict__ ex, const float* __restrict__ den,
                        const float* __restrict__ v, const int* src, const int* dst,
                        float* __restrict__ base, int total) {
  int idx = blockIdx.x * blockDim.x + threadIdx.x;
  if (idx >= total) return;                 // total = EE*16
  int chunk = idx & 15;
  int e = idx >> 4;
  float a = ex[e] / den[dst[e]];
  const float* vr = v + (long)src[e] * HH + chunk * 8;
  float* br = base + (long)dst[e] * HH + chunk * 8;
  #pragma unroll
  for (int j = 0; j < 8; ++j) atomicAdd(&br[j], a * vr[j]);
}
__global__ void k_lnrelu(const float* __restrict__ in, const float* g, const float* b,
                         float* __restrict__ out, int n) {
  int i = blockIdx.x * blockDim.x + threadIdx.x;
  if (i >= n) return;
  const float* xr = in + (long)i * HH;
  float mu = 0.f;
  for (int f = 0; f < HH; ++f) mu += xr[f];
  mu *= (1.f / HH);
  float var = 0.f;
  for (int f = 0; f < HH; ++f) { float d = xr[f] - mu; var += d * d; }
  var *= (1.f / HH);
  float is = rsqrtf(var + 1e-5f);
  float* orow = out + (long)i * HH;
  for (int f = 0; f < HH; ++f) orow[f] = fmaxf((xr[f] - mu) * is * g[f] + b[f], 0.f);
}
__global__ void k_concat(const float* tb, const float* hif, float* hc, int total) {
  int i = blockIdx.x * blockDim.x + threadIdx.x;
  if (i >= total) return;                   // total = N*2H
  int n = i >> 8, f = i & 255;
  hc[i] = (f < HH) ? tb[(long)n * HH + f] : hif[(long)n * HH + f - HH];
}
__global__ void k_f2h(const float* in, _Float16* out, int n) {
  int i = blockIdx.x * blockDim.x + threadIdx.x;
  if (i < n) out[i] = (_Float16)in[i];
}

// ---------------------------------------------------------------------------
extern "C" void kernel_launch(void* const* d_in, const int* in_sizes, int n_in,
                              void* d_out, int out_size, void* d_ws, size_t ws_size,
                              hipStream_t stream) {
  const float* x   = (const float*)d_in[0];
  const int*   ei  = (const int*)d_in[1];
  const int*   src = ei;
  const int*   dst = ei + EE;
  const float* ew  = (const float*)d_in[2];
  // params (dict order)
  const float* Wc_z = (const float*)d_in[3];  const float* bc_z = (const float*)d_in[4];
  const float* Wl_z = (const float*)d_in[5];  const float* bl_z = (const float*)d_in[6];
  // 7..10: r-gate weights unused (H0 == 0 => r is dead)
  const float* Wc_h = (const float*)d_in[11]; const float* bc_h = (const float*)d_in[12];
  const float* Wl_h = (const float*)d_in[13]; const float* bl_h = (const float*)d_in[14];
  const float* Wp    = (const float*)d_in[31]; const float* bp    = (const float*)d_in[32];
  const float* Wconv = (const float*)d_in[33]; const float* bconv = (const float*)d_in[34];
  const float* Wskip = (const float*)d_in[35]; const float* bskip = (const float*)d_in[36];
  const float* g1 = (const float*)d_in[37]; const float* be1 = (const float*)d_in[38];
  const float* g2 = (const float*)d_in[39]; const float* be2 = (const float*)d_in[40];
  const float* Wn  = (const float*)d_in[41]; const float* bn  = (const float*)d_in[42];
  const float* Wi1 = (const float*)d_in[43]; const float* bi1 = (const float*)d_in[44];
  const float* Wi2 = (const float*)d_in[45]; const float* bi2 = (const float*)d_in[46];
  const float* Wi3 = (const float*)d_in[47]; const float* bi3 = (const float*)d_in[48];

  float* out_np    = (float*)d_out;                 // [N, 32]
  float* out_inter = (float*)d_out + NND * OUTC;    // [N, N]

  // workspace layout
  float* ws = (float*)d_ws;
  size_t o = 0;
  auto AL = [&](size_t n) { float* p = ws + o; o += n; return p; };
  const size_t BIG = (size_t)SEQ * NND * HH;        // 2,097,152
  float* xwz  = AL(BIG);
  float* xwh  = AL(BIG);
  float* cz   = AL(BIG);
  float* chb  = AL(BIG);
  float* hseq = AL(BIG);
  float* Zb = xwz;                                   // reuse after scatter
  float* Hb = xwh;
  float* deg = AL(NND); float* dis = AL(NND); float* nrm = AL(EE);
  float* mrow = AL(NND); float* den = AL(NND);
  float* sc = AL(EE); float* exb = AL(EE);
  float* hmean = AL(NND * HH);
  float* av = AL(NND * HH); float* a0 = AL(NND * HH); float* aL = AL(NND * HH);
  float* tb = AL(NND * HH);
  float* Wsum = AL(HH * HH); float* W0b = AL(HH * HH); float* W2b = AL(HH * HH);
  float* qb = AL(NND * HH); float* kb = AL(NND * HH); float* vb = AL(NND * HH);
  float* base = AL(NND * HH);
  float* hi1 = AL(NND * HH); float* hi2 = AL(NND * HH); float* hif = AL(NND * HH);
  float* hcb = AL(NND * 2 * HH);
  float* zab = AL(NND * HH); float* zbb = AL(NND * HH);
  _Float16* w2h = (_Float16*)AL((64 * HH) / 2);

  const int MT = SEQ * NND;                          // 16384 rows for all timesteps

  // ---- GCN normalization ----
  k_fill<<<CDIV(NND, 256), 256, 0, stream>>>(deg, 1.0f, NND);     // self-loop weight 1
  k_deg<<<CDIV(EE, 256), 256, 0, stream>>>(dst, ew, deg, EE);
  k_dis<<<CDIV(NND, 256), 256, 0, stream>>>(deg, dis, NND);
  k_norm<<<CDIV(EE, 256), 256, 0, stream>>>(src, dst, ew, dis, nrm, EE);

  // ---- TGCN (all 16 steps batched): xw = x @ Wc^T ----
  gemm(stream, x, FIN, Wc_z, FIN, nullptr, nullptr, 0, xwz, HH, MT, HH, FIN);
  gemm(stream, x, FIN, Wc_h, FIN, nullptr, nullptr, 0, xwh, HH, MT, HH, FIN);
  k_gcn_init<<<CDIV((int)BIG, 256), 256, 0, stream>>>(xwz, dis, bc_z, cz, (int)BIG);
  k_gcn_init<<<CDIV((int)BIG, 256), 256, 0, stream>>>(xwh, dis, bc_h, chb, (int)BIG);
  const int SCT = SEQ * EE * 16;
  k_gcn_scatter<<<CDIV(SCT, 256), 256, 0, stream>>>(xwz, src, dst, nrm, cz, SCT);
  k_gcn_scatter<<<CDIV(SCT, 256), 256, 0, stream>>>(xwh, src, dst, nrm, chb, SCT);
  // gates: only first H columns of Wl_* matter (H0 == 0)
  gemm(stream, cz, HH, Wl_z, 2 * HH, bl_z, nullptr, 0, Zb, HH, MT, HH, HH, 1.f, 1);  // sigmoid
  gemm(stream, chb, HH, Wl_h, 2 * HH, bl_h, nullptr, 0, Hb, HH, MT, HH, HH, 1.f, 2); // tanh
  k_gates<<<CDIV((int)BIG, 256), 256, 0, stream>>>(Zb, Hb, hseq, (int)BIG);

  // ---- temporal branch (Conv1d+mean folded into 3 GEMMs) ----
  k_meantime<<<CDIV(NND * HH, 256), 256, 0, stream>>>(hseq, hmean, NND * HH);
  gemm(stream, hmean, HH, Wp, HH, bp, nullptr, 0, av, HH, NND, HH, HH);
  gemm(stream, hseq, HH, Wp, HH, bp, nullptr, 0, a0, HH, NND, HH, HH);                       // h[0]
  gemm(stream, hseq + (long)(SEQ - 1) * NND * HH, HH, Wp, HH, bp, nullptr, 0, aL, HH, NND, HH, HH); // h[S-1]
  k_unpack_conv<<<CDIV(HH * HH, 256), 256, 0, stream>>>(Wconv, Wsum, W0b, W2b, HH * HH);
  gemm(stream, av, HH, Wsum, HH, bconv, nullptr, 0, tb, HH, NND, HH, HH);
  gemm(stream, aL, HH, W0b, HH, nullptr, tb, HH, tb, HH, NND, HH, HH, -1.f / SEQ);
  gemm(stream, a0, HH, W2b, HH, nullptr, tb, HH, tb, HH, NND, HH, HH, -1.f / SEQ);

  // ---- interaction branch: two TransformerConv layers ----
  auto tconv = [&](const float* hin, int pb, const float* g, const float* be, float* hout) {
    const float* Wq = (const float*)d_in[pb + 0]; const float* bq = (const float*)d_in[pb + 1];
    const float* Wk = (const float*)d_in[pb + 2]; const float* bk = (const float*)d_in[pb + 3];
    const float* Wv = (const float*)d_in[pb + 4]; const float* bv = (const float*)d_in[pb + 5];
    const float* Ws = (const float*)d_in[pb + 6]; const float* bs = (const float*)d_in[pb + 7];
    gemm(stream, hin, HH, Wq, HH, bq, nullptr, 0, qb, HH, NND, HH, HH);
    gemm(stream, hin, HH, Wk, HH, bk, nullptr, 0, kb, HH, NND, HH, HH);
    gemm(stream, hin, HH, Wv, HH, bv, nullptr, 0, vb, HH, NND, HH, HH);
    gemm(stream, hin, HH, Ws, HH, bs, nullptr, 0, base, HH, NND, HH, HH);  // root skip
    k_scores<<<CDIV(EE, 256), 256, 0, stream>>>(qb, kb, src, dst, sc, EE);
    k_fill<<<CDIV(NND, 256), 256, 0, stream>>>(mrow, -3.0e38f, NND);
    k_fill<<<CDIV(NND, 256), 256, 0, stream>>>(den, 0.f, NND);
    k_segmax<<<CDIV(EE, 256), 256, 0, stream>>>(sc, dst, mrow, EE);
    k_expden<<<CDIV(EE, 256), 256, 0, stream>>>(sc, dst, mrow, exb, den, EE);
    k_attsc<<<CDIV(EE * 16, 256), 256, 0, stream>>>(exb, den, vb, src, dst, base, EE * 16);
    k_lnrelu<<<CDIV(NND, 128), 128, 0, stream>>>(base, g, be, hout, NND);
  };
  tconv(hmean, 15, g1, be1, hi1);   // t1
  tconv(hi1, 23, g2, be2, hi2);     // t2
  gemm(stream, hi2, HH, Wskip, HH, bskip, hi2, HH, hif, HH, NND, HH, HH);  // hi + skip

  // ---- heads ----
  k_concat<<<CDIV(NND * 2 * HH, 256), 256, 0, stream>>>(tb, hif, hcb, NND * 2 * HH);
  gemm(stream, hcb, 2 * HH, Wn, 2 * HH, bn, nullptr, 0, out_np, OUTC, NND, OUTC, 2 * HH);
  gemm(stream, hcb, 2 * HH, Wi1, 4 * HH, nullptr, nullptr, 0, zab, HH, NND, HH, 2 * HH);      // Wa
  gemm(stream, hcb, 2 * HH, Wi1 + 2 * HH, 4 * HH, nullptr, nullptr, 0, zbb, HH, NND, HH, 2 * HH); // Wb
  k_f2h<<<CDIV(64 * HH, 256), 256, 0, stream>>>(Wi2, w2h, 64 * HH);

  // ---- pairwise interaction (dominant: ~17 GFLOP, fully WMMA) ----
  dim3 pg(NND / 16, NND);
  k_pairwise<<<pg, 32, 0, stream>>>(zab, zbb, bi1, w2h, bi2, Wi3, bi3, out_inter);
  (void)in_sizes; (void)n_in; (void)out_size; (void)ws_size;
}
